// GIN_74406013436496
// MI455X (gfx1250) — matrix operand order repaired and checked by reading, not compiled, not run on hardware
//
#include <hip/hip_runtime.h>
#include <hip/hip_bf16.h>
#include <math.h>

typedef __bf16 bf16_t;
typedef __attribute__((ext_vector_type(16))) __bf16 v16bf;
typedef __attribute__((ext_vector_type(8)))  __bf16 v8bf;
typedef __attribute__((ext_vector_type(8)))  float  v8f;

#define HFEAT 128
#define NUM_GRAPHS 1024
#define NUM_CLASSES 10
#define NEG_SLOPE 0.01f
#define BN_EPS 1e-5f

// XOR-swizzled LDS addressing for 128-wide bf16 tiles: 8-element groups stay
// contiguous (16B-aligned for ds_load_b128), banks decorrelate across rows.
__device__ __forceinline__ int swz(int row, int k) {
    return (row << 7) | ((((k >> 3) ^ row) & 15) << 3) | (k & 7);
}

__device__ __forceinline__ float leaky(float v) {
    return v > 0.0f ? v : NEG_SLOPE * v;
}

// ---------------------------------------------------------------------------
// Fused GINConv MLP: out = leaky( (leaky(in @ Wa + ba)) @ Wb + bb )
// One block = 128 rows x 128 cols, 8 waves, each wave owns a 16-row strip.
// bf16 WMMA (16x16x32), f32 accumulate. Mid matrix stored back into sA
// in-place (wave-private rows), W^T staged in sW (swizzled).
// ---------------------------------------------------------------------------
__global__ __launch_bounds__(256)
void gin_conv_kernel(const float* __restrict__ in,
                     const float* __restrict__ Wa, const float* __restrict__ ba,
                     const float* __restrict__ Wb, const float* __restrict__ bb,
                     float* __restrict__ out, int N)
{
    __shared__ bf16_t sA[128 * 128];  // input tile -> (in-place) mid tile
    __shared__ bf16_t sW[128 * 128];  // W^T, swizzled: sW[n][k]

    const int t    = threadIdx.x;
    const int lane = t & 31;
    const int wave = t >> 5;
    const int rowBase = blockIdx.x * 128;

    // hint: second weight matrix will be needed soon
    if (t < 64) __builtin_prefetch(&Wb[t * 256], 0, 0);

    // Stage input tile fp32 -> bf16 (coalesced float4 reads), zero-pad tail.
    #pragma unroll
    for (int i = 0; i < 16; ++i) {
        int idx4 = t + i * 256;            // float4 index in 128x128 tile
        int r  = idx4 >> 5;
        int c4 = (idx4 & 31) << 2;
        float4 v = make_float4(0.f, 0.f, 0.f, 0.f);
        int gr = rowBase + r;
        if (gr < N) v = *(const float4*)(in + (size_t)gr * HFEAT + c4);
        bf16_t* p = &sA[swz(r, c4)];
        p[0] = (bf16_t)v.x; p[1] = (bf16_t)v.y;
        p[2] = (bf16_t)v.z; p[3] = (bf16_t)v.w;
    }
    // Stage Wa transposed: sW[n][k] = Wa[k*128 + n]
    #pragma unroll
    for (int i = 0; i < 64; ++i) {
        int idx = t + i * 256;
        int k = idx >> 7, n = idx & 127;
        sW[swz(n, k)] = (bf16_t)Wa[idx];
    }
    __syncthreads();

    const int hl    = lane >> 4;           // half-wave 0/1
    const int l15   = lane & 15;
    const int arow  = (wave << 4) + l15;   // A-fragment row (M)
    const int mrow0 = (wave << 4) + (hl << 3);  // C/D fragment row base

    const v8f vzero = {0.f, 0.f, 0.f, 0.f, 0.f, 0.f, 0.f, 0.f};
    v8f acc[8];
    #pragma unroll
    for (int i = 0; i < 8; ++i) acc[i] = vzero;

    // ---- GEMM 1: sA(128x128) @ Wa(128x128) ----
    #pragma unroll
    for (int kc = 0; kc < 4; ++kc) {
        // A 16x32 bf16 layout: lanes 0-15 K runs [kc*32+0..7, +16..23],
        // lanes 16-31 runs [+8..15, +24..31]
        int b0 = (kc << 5) + (hl << 3);
        v8bf alo = *(const v8bf*)&sA[swz(arow, b0)];
        v8bf ahi = *(const v8bf*)&sA[swz(arow, b0 + 16)];
        v16bf af = __builtin_shufflevector(alo, ahi,
            0,1,2,3,4,5,6,7,8,9,10,11,12,13,14,15);
        int koff = (kc << 5) + (hl << 3);  // B: same K runs as A
        #pragma unroll
        for (int nt = 0; nt < 8; ++nt) {
            int bcol = (nt << 4) + l15;
            v8bf blo = *(const v8bf*)&sW[swz(bcol, koff)];
            v8bf bhi = *(const v8bf*)&sW[swz(bcol, koff + 16)];
            v16bf bfrag = __builtin_shufflevector(blo, bhi,
                0,1,2,3,4,5,6,7,8,9,10,11,12,13,14,15);
            acc[nt] = __builtin_amdgcn_wmma_f32_16x16x32_bf16(
                false, af, false, bfrag, (short)0, acc[nt], false, false);
        }
    }

    // Epilogue 1: bias + leaky, write mid back into sA (wave-private rows).
    #pragma unroll
    for (int nt = 0; nt < 8; ++nt) {
        int col = (nt << 4) + l15;
        float bv = ba[col];
        #pragma unroll
        for (int r = 0; r < 8; ++r) {
            float v = leaky(acc[nt][r] + bv);
            sA[swz(mrow0 + r, col)] = (bf16_t)v;
        }
    }
    __syncthreads();

    // Stage Wb transposed (sW reuse; all waves past GEMM1 reads).
    #pragma unroll
    for (int i = 0; i < 64; ++i) {
        int idx = t + i * 256;
        int k = idx >> 7, n = idx & 127;
        sW[swz(n, k)] = (bf16_t)Wb[idx];
    }
    __syncthreads();

    #pragma unroll
    for (int i = 0; i < 8; ++i) acc[i] = vzero;

    // ---- GEMM 2: mid(128x128) @ Wb(128x128) ----
    #pragma unroll
    for (int kc = 0; kc < 4; ++kc) {
        int b0 = (kc << 5) + (hl << 3);
        v8bf alo = *(const v8bf*)&sA[swz(arow, b0)];
        v8bf ahi = *(const v8bf*)&sA[swz(arow, b0 + 16)];
        v16bf af = __builtin_shufflevector(alo, ahi,
            0,1,2,3,4,5,6,7,8,9,10,11,12,13,14,15);
        int koff = (kc << 5) + (hl << 3);
        #pragma unroll
        for (int nt = 0; nt < 8; ++nt) {
            int bcol = (nt << 4) + l15;
            v8bf blo = *(const v8bf*)&sW[swz(bcol, koff)];
            v8bf bhi = *(const v8bf*)&sW[swz(bcol, koff + 16)];
            v16bf bfrag = __builtin_shufflevector(blo, bhi,
                0,1,2,3,4,5,6,7,8,9,10,11,12,13,14,15);
            acc[nt] = __builtin_amdgcn_wmma_f32_16x16x32_bf16(
                false, af, false, bfrag, (short)0, acc[nt], false, false);
        }
    }

    // Epilogue 2: bias + outer leaky, fp32 store to global.
    #pragma unroll
    for (int nt = 0; nt < 8; ++nt) {
        int col = (nt << 4) + l15;
        float bv = bb[col];
        #pragma unroll
        for (int r = 0; r < 8; ++r) {
            int gr = rowBase + mrow0 + r;
            if (gr < N)
                out[(size_t)gr * HFEAT + col] = leaky(acc[nt][r] + bv);
        }
    }
}

// ---------------------------------------------------------------------------
// Edge scatter-add: dest[dst] += src_feat[src]. One wave per edge,
// float4 per lane (32 lanes x 4 = 128 feats), f32 global atomics.
// ---------------------------------------------------------------------------
__global__ __launch_bounds__(256)
void scatter_add_kernel(const float* __restrict__ src_feat,
                        float* __restrict__ dst_feat,
                        const int* __restrict__ ei, int E)
{
    int e = blockIdx.x * 8 + (threadIdx.x >> 5);
    if (e >= E) return;
    int lane = threadIdx.x & 31;
    int s = ei[e];
    int d = ei[E + e];
    float4 v = *(const float4*)(src_feat + (size_t)s * HFEAT + lane * 4);
    float* p = dst_feat + (size_t)d * HFEAT + lane * 4;
    atomicAdd(p + 0, v.x);
    atomicAdd(p + 1, v.y);
    atomicAdd(p + 2, v.z);
    atomicAdd(p + 3, v.w);
}

// ---------------------------------------------------------------------------
// BatchNorm: per-feature partial sum/sumsq, atomically folded into stats.
// ---------------------------------------------------------------------------
__global__ __launch_bounds__(256)
void bn_stats_kernel(const float* __restrict__ h, float* __restrict__ sums,
                     float* __restrict__ sqs, int N)
{
    const int f    = threadIdx.x & 127;
    const int half = threadIdx.x >> 7;
    int r0   = blockIdx.x * 256;
    int rend = r0 + 256; if (rend > N) rend = N;
    float s = 0.f, q = 0.f;
    for (int r = r0 + half; r < rend; r += 2) {
        float v = h[(size_t)r * HFEAT + f];
        s += v; q += v * v;
    }
    __shared__ float shs[256], shq[256];
    shs[threadIdx.x] = s; shq[threadIdx.x] = q;
    __syncthreads();
    if (half == 0) {
        atomicAdd(&sums[f], shs[f] + shs[128 + f]);
        atomicAdd(&sqs[f],  shq[f] + shq[128 + f]);
    }
}

__global__ __launch_bounds__(128)
void bn_final_kernel(const float* __restrict__ sums, const float* __restrict__ sqs,
                     const float* __restrict__ gamma, const float* __restrict__ beta,
                     float* __restrict__ a, float* __restrict__ b, int N)
{
    int f = threadIdx.x;
    float inv  = 1.0f / (float)N;
    float m    = sums[f] * inv;
    float var  = sqs[f] * inv - m * m;
    float rstd = rsqrtf(var + BN_EPS);
    float av   = gamma[f] * rstd;
    a[f] = av;
    b[f] = beta[f] - m * av;
}

__global__ __launch_bounds__(256)
void bn_apply_kernel(const float4* __restrict__ in, float4* __restrict__ out,
                     const float* __restrict__ a, const float* __restrict__ b, int n4)
{
    int i = blockIdx.x * 256 + threadIdx.x;
    if (i >= n4) return;
    int f = (i << 2) & 127;
    float4 v = in[i];
    float4 r;
    r.x = a[f + 0] * v.x + b[f + 0];
    r.y = a[f + 1] * v.y + b[f + 1];
    r.z = a[f + 2] * v.z + b[f + 2];
    r.w = a[f + 3] * v.w + b[f + 3];
    out[i] = r;
}

// ---------------------------------------------------------------------------
// global_add_pool: hg[batch[r]] += h[r]
// ---------------------------------------------------------------------------
__global__ __launch_bounds__(256)
void pool_kernel(const float* __restrict__ h, const int* __restrict__ batch,
                 float* __restrict__ hg, int total)
{
    int idx = blockIdx.x * 256 + threadIdx.x;
    if (idx >= total) return;
    int r = idx >> 7;
    int f = idx & 127;
    int g = batch[r];
    atomicAdd(&hg[(size_t)g * HFEAT + f], h[idx]);
}

// ---------------------------------------------------------------------------
// MLP head + log_softmax: one 128-thread block per graph (tiny, latency-bound).
// ---------------------------------------------------------------------------
__global__ __launch_bounds__(128)
void mlp_head_kernel(const float* __restrict__ hg,
                     const float* __restrict__ W1, const float* __restrict__ b1,
                     const float* __restrict__ W2, const float* __restrict__ b2,
                     float* __restrict__ out)
{
    __shared__ float sh[128];
    __shared__ float sz[128];
    __shared__ float sl[NUM_CLASSES];
    __shared__ float slse;
    const int t = threadIdx.x;
    const int g = blockIdx.x;

    sh[t] = hg[(size_t)g * HFEAT + t];
    __syncthreads();

    float acc = b1[t];
    #pragma unroll 8
    for (int k = 0; k < 128; ++k) acc += sh[k] * W1[k * 128 + t];
    sz[t] = leaky(acc);
    __syncthreads();

    if (t < NUM_CLASSES) {
        float l = b2[t];
        #pragma unroll 8
        for (int k = 0; k < 128; ++k) l += sz[k] * W2[k * NUM_CLASSES + t];
        sl[t] = l;
    }
    __syncthreads();

    if (t == 0) {
        float m = sl[0];
        #pragma unroll
        for (int i = 1; i < NUM_CLASSES; ++i) m = fmaxf(m, sl[i]);
        float s = 0.f;
        #pragma unroll
        for (int i = 0; i < NUM_CLASSES; ++i) s += expf(sl[i] - m);
        slse = m + logf(s);
    }
    __syncthreads();

    if (t < NUM_CLASSES) out[(size_t)g * NUM_CLASSES + t] = sl[t] - slse;
}

// ---------------------------------------------------------------------------
extern "C" void kernel_launch(void* const* d_in, const int* in_sizes, int n_in,
                              void* d_out, int out_size, void* d_ws, size_t ws_size,
                              hipStream_t stream)
{
    const int N = in_sizes[0] / HFEAT;
    const int E = in_sizes[17] / 2;

    const float* x    = (const float*)d_in[0];
    const float* W1a  = (const float*)d_in[1];
    const float* b1a  = (const float*)d_in[2];
    const float* W1b  = (const float*)d_in[3];
    const float* b1b  = (const float*)d_in[4];
    const float* g1   = (const float*)d_in[5];
    const float* be1  = (const float*)d_in[6];
    const float* W2a  = (const float*)d_in[7];
    const float* b2a  = (const float*)d_in[8];
    const float* W2b  = (const float*)d_in[9];
    const float* b2b  = (const float*)d_in[10];
    const float* g2   = (const float*)d_in[11];
    const float* be2  = (const float*)d_in[12];
    const float* fcW1 = (const float*)d_in[13];
    const float* fcb1 = (const float*)d_in[14];
    const float* fcW2 = (const float*)d_in[15];
    const float* fcb2 = (const float*)d_in[16];
    const int* ei     = (const int*)d_in[17];
    const int* batch  = (const int*)d_in[18];
    float* out        = (float*)d_out;

    const size_t nodeElems = (size_t)N * HFEAT;
    float* ws    = (float*)d_ws;
    float* buf0  = ws;
    float* buf1  = ws + nodeElems;
    float* hg    = ws + 2 * nodeElems;
    float* stats = hg + (size_t)NUM_GRAPHS * HFEAT;
    float* sums1 = stats;       float* sqs1 = stats + 128;
    float* a1c   = stats + 256; float* b1c  = stats + 384;
    float* sums2 = stats + 512; float* sqs2 = stats + 640;
    float* a2c   = stats + 768; float* b2c  = stats + 896;

    const int scatterBlocks = (E + 7) / 8;
    const int gemmBlocks    = (N + 127) / 128;
    const int statBlocks    = (N + 255) / 256;
    const int n4            = (int)(nodeElems / 4);
    const int applyBlocks   = (n4 + 255) / 256;
    const int poolTotal     = N * HFEAT;
    const int poolBlocks    = (poolTotal + 255) / 256;

    // Zero pool accumulators + BN stats (graph-capture safe).
    hipMemsetAsync(hg, 0, ((size_t)NUM_GRAPHS * HFEAT + 1024) * sizeof(float), stream);

    // ---- Layer 1 ----
    hipMemcpyAsync(buf0, x, nodeElems * sizeof(float), hipMemcpyDeviceToDevice, stream);
    scatter_add_kernel<<<scatterBlocks, 256, 0, stream>>>(x, buf0, ei, E);
    gin_conv_kernel<<<gemmBlocks, 256, 0, stream>>>(buf0, W1a, b1a, W1b, b1b, buf1, N);
    bn_stats_kernel<<<statBlocks, 256, 0, stream>>>(buf1, sums1, sqs1, N);
    bn_final_kernel<<<1, 128, 0, stream>>>(sums1, sqs1, g1, be1, a1c, b1c, N);
    bn_apply_kernel<<<applyBlocks, 256, 0, stream>>>((const float4*)buf1, (float4*)buf0,
                                                     a1c, b1c, n4);

    // ---- Layer 2 ----
    hipMemcpyAsync(buf1, buf0, nodeElems * sizeof(float), hipMemcpyDeviceToDevice, stream);
    scatter_add_kernel<<<scatterBlocks, 256, 0, stream>>>(buf0, buf1, ei, E);
    gin_conv_kernel<<<gemmBlocks, 256, 0, stream>>>(buf1, W2a, b2a, W2b, b2b, buf1, N);
    bn_stats_kernel<<<statBlocks, 256, 0, stream>>>(buf1, sums2, sqs2, N);
    bn_final_kernel<<<1, 128, 0, stream>>>(sums2, sqs2, g2, be2, a2c, b2c, N);
    bn_apply_kernel<<<applyBlocks, 256, 0, stream>>>((const float4*)buf1, (float4*)buf0,
                                                     a2c, b2c, n4);

    // ---- Pool + head ----
    pool_kernel<<<poolBlocks, 256, 0, stream>>>(buf0, batch, hg, poolTotal);
    mlp_head_kernel<<<NUM_GRAPHS, 128, 0, stream>>>(hg, fcW1, fcb1, fcW2, fcb2, out);

    (void)n_in; (void)out_size; (void)ws_size;
}
